// HyperExplainer_17437567221911
// MI455X (gfx1250) — matrix operand order repaired and checked by reading, not compiled, not run on hardware
//
#include <hip/hip_runtime.h>
#include <hip/hip_bf16.h>

#define D_DIM 256
#define H_DIM 16
#define TOPK  5

typedef __attribute__((ext_vector_type(16))) __bf16 v16bf;
typedef __attribute__((ext_vector_type(8)))  float  v8f;

// f32 -> bf16, round to nearest even
__device__ __forceinline__ __bf16 f2bf(float f) {
    unsigned u = __float_as_uint(f);
    unsigned r = u + 0x7FFFu + ((u >> 16) & 1u);
    unsigned short h = (unsigned short)(r >> 16);
    return __builtin_bit_cast(__bf16, h);
}

// ---------------- init: zero seg_sum+cnt, reset selection state ----------------
__global__ void k_init(float* seg_and_cnt, unsigned long long* cell, long long* sel,
                       long long total) {
    long long i = (long long)blockIdx.x * blockDim.x + threadIdx.x;
    long long stride = (long long)gridDim.x * blockDim.x;
    for (long long j = i; j < total; j += stride) seg_and_cnt[j] = 0.0f;
    if (i == 0) {
        *cell = 0ull;
        for (int k = 0; k < TOPK; ++k) sel[k] = -1;
    }
}

// ---------------- segment scatter-add: seg[h] += x[n], cnt[h] += 1 ----------------
__global__ void k_scatter(const float* __restrict__ x, const int* __restrict__ en,
                          const int* __restrict__ eh, float* __restrict__ seg,
                          float* __restrict__ cnt, int E) {
    long long total = (long long)E * (D_DIM / 4);
    long long i = (long long)blockIdx.x * blockDim.x + threadIdx.x;
    long long stride = (long long)gridDim.x * blockDim.x;
    for (long long j = i; j < total; j += stride) {
        int e = (int)(j >> 6);          // D/4 == 64 chunks per edge
        int q = (int)(j & 63);
        int n = en[e];
        int h = eh[e];
        float4 v = ((const float4*)x)[(long long)n * (D_DIM / 4) + q];
        float* dst = seg + (long long)h * D_DIM + q * 4;
        __hip_atomic_fetch_add(dst + 0, v.x, __ATOMIC_RELAXED, __HIP_MEMORY_SCOPE_AGENT);
        __hip_atomic_fetch_add(dst + 1, v.y, __ATOMIC_RELAXED, __HIP_MEMORY_SCOPE_AGENT);
        __hip_atomic_fetch_add(dst + 2, v.z, __ATOMIC_RELAXED, __HIP_MEMORY_SCOPE_AGENT);
        __hip_atomic_fetch_add(dst + 3, v.w, __ATOMIC_RELAXED, __HIP_MEMORY_SCOPE_AGENT);
        if (q == 0)
            __hip_atomic_fetch_add(cnt + h, 1.0f, __ATOMIC_RELAXED, __HIP_MEMORY_SCOPE_AGENT);
    }
}

// ---------------- relu(A @ W + b) via v_wmma_f32_16x16x32_bf16 ----------------
// A: [rows,256] f32 (optionally divided per-row by max(cnt,1)), W: [256,16], out: [rows,16]
// One wave32 per 16-row tile; rows must be a multiple of 16 (100000 and 50000 are).
__global__ void k_mlp_wmma(const float* __restrict__ A, const float* __restrict__ cnt,
                           const float* __restrict__ W, const float* __restrict__ bvec,
                           float* __restrict__ Hout, int nTiles) {
    const int lane  = threadIdx.x & 31;
    const int wave  = threadIdx.x >> 5;
    const int wpb   = blockDim.x >> 5;
    const int gwave = blockIdx.x * wpb + wave;
    const int nW    = gridDim.x * wpb;
    const int col   = lane & 15;   // C/D column, B column, A row-within-tile
    const int kh    = lane >> 4;   // which K-half this lane owns

    // Preload B (weights) once per wave in WMMA wave32 layout.
    // B 32x16 bf16 chunk kc: lane(0-15)=col holds K = kc*32 + kh*16 + j  (j = element 0..15)
    v16bf Bm[8];
#pragma unroll
    for (int kc = 0; kc < 8; ++kc) {
        const int kbase = kc * 32 + kh * 16;
#pragma unroll
        for (int j = 0; j < 16; ++j)
            Bm[kc][j] = f2bf(W[(kbase + j) * H_DIM + col]);
    }

    for (int tile = gwave; tile < nTiles; tile += nW) {
        const int row = tile * 16 + col;   // A-matrix row this lane loads
        float scale = 1.0f;
        if (cnt) scale = 1.0f / fmaxf(cnt[row], 1.0f);
        const float* arow = A + (long long)row * D_DIM;

        v8f acc = {};
#pragma unroll
        for (int kc = 0; kc < 8; ++kc) {
            const int k0 = kc * 32;
            // A 16x32 bf16: lane 0-15 -> e0..15 = K k0..k0+15
            //               lane16-31 -> e0..15 = K k0+16..k0+31
            float4 p0 = *(const float4*)(arow + k0 + kh * 16);
            float4 p1 = *(const float4*)(arow + k0 + kh * 16 + 4);
            float4 p2 = *(const float4*)(arow + k0 + kh * 16 + 8);
            float4 p3 = *(const float4*)(arow + k0 + kh * 16 + 12);
            v16bf a;
            a[0]  = f2bf(p0.x * scale); a[1]  = f2bf(p0.y * scale);
            a[2]  = f2bf(p0.z * scale); a[3]  = f2bf(p0.w * scale);
            a[4]  = f2bf(p1.x * scale); a[5]  = f2bf(p1.y * scale);
            a[6]  = f2bf(p1.z * scale); a[7]  = f2bf(p1.w * scale);
            a[8]  = f2bf(p2.x * scale); a[9]  = f2bf(p2.y * scale);
            a[10] = f2bf(p2.z * scale); a[11] = f2bf(p2.w * scale);
            a[12] = f2bf(p3.x * scale); a[13] = f2bf(p3.y * scale);
            a[14] = f2bf(p3.z * scale); a[15] = f2bf(p3.w * scale);
            acc = __builtin_amdgcn_wmma_f32_16x16x32_bf16(
                false, a, false, Bm[kc], (short)0, acc, false, false);
        }

        // C/D 16x16 f32: VGPR r -> row (kh*8 + r), lane%16 -> col
        const float bb = bvec[col];
#pragma unroll
        for (int r = 0; r < 8; ++r) {
            int m = tile * 16 + kh * 8 + r;
            float v = acc[r] + bb;
            Hout[(long long)m * H_DIM + col] = v > 0.0f ? v : 0.0f;
        }
    }
}

// ---------------- per-incidence attention scores ----------------
__global__ void k_scores(const int* __restrict__ en, const int* __restrict__ eh,
                         const float* __restrict__ hn, const float* __restrict__ he,
                         const float* __restrict__ bias, const int* __restrict__ node_id,
                         float* __restrict__ out, int E) {
    int i = blockIdx.x * blockDim.x + threadIdx.x;
    int stride = gridDim.x * blockDim.x;
    float b = bias[0];
    int nid = node_id[0];
    for (; i < E; i += stride) {
        int n = en[i], h = eh[i];
        const float4* pn = (const float4*)(hn + (long long)n * H_DIM);
        const float4* ph = (const float4*)(he + (long long)h * H_DIM);
        float s = 0.0f;
#pragma unroll
        for (int q = 0; q < 4; ++q) {
            float4 a = pn[q], c = ph[q];
            s += a.x * c.x + a.y * c.y + a.z * c.z + a.w * c.w;
        }
        float w = (n == nid) ? 1.0f : b;
        out[i] = s * w;
    }
}

// ---------------- top-K selection: argmax pass excluding prior picks ----------------
__device__ __forceinline__ unsigned long long fkey(float f, unsigned idx) {
    unsigned b = __float_as_uint(f);
    unsigned o = (b & 0x80000000u) ? ~b : (b | 0x80000000u);   // order-preserving map
    // invert index so ties pick the SMALLEST index (matches jax top_k tie-break)
    return ((unsigned long long)o << 32) | (unsigned long long)(0xFFFFFFFFu - idx);
}

__global__ void k_argmax(const float* __restrict__ s, const long long* __restrict__ sel,
                         unsigned long long* cell, int E) {
    long long s0 = sel[0], s1 = sel[1], s2 = sel[2], s3 = sel[3], s4 = sel[4];
    unsigned long long best = 0ull;
    int i = blockIdx.x * blockDim.x + threadIdx.x;
    int stride = gridDim.x * blockDim.x;
    for (; i < E; i += stride) {
        long long li = i;
        if (li == s0 || li == s1 || li == s2 || li == s3 || li == s4) continue;
        unsigned long long k = fkey(s[i], (unsigned)i);
        if (k > best) best = k;
    }
#pragma unroll
    for (int off = 16; off > 0; off >>= 1) {
        unsigned long long o = __shfl_xor(best, off, 32);
        if (o > best) best = o;
    }
    if ((threadIdx.x & 31) == 0 && best)
        atomicMax(cell, best);
}

__global__ void k_record(unsigned long long* cell, long long* sel, int k) {
    unsigned idx = 0xFFFFFFFFu - (unsigned)(*cell & 0xFFFFFFFFull);
    sel[k] = (long long)idx;
    *cell = 0ull;   // ready for next pass
}

__global__ void k_prune(const long long* __restrict__ sel, float* __restrict__ out, int E) {
    int k = threadIdx.x;
    if (k < TOPK) {
        long long i = sel[k];
        if (i >= 0 && i < E) out[i] = 0.0f;
    }
}

extern "C" void kernel_launch(void* const* d_in, const int* in_sizes, int n_in,
                              void* d_out, int out_size, void* d_ws, size_t ws_size,
                              hipStream_t stream) {
    const float* x    = (const float*)d_in[0];
    const float* W_n  = (const float*)d_in[1];
    const float* b_n  = (const float*)d_in[2];
    const float* W_e  = (const float*)d_in[3];
    const float* b_e  = (const float*)d_in[4];
    const float* bias = (const float*)d_in[5];
    const int*   en   = (const int*)d_in[6];
    const int*   eh   = (const int*)d_in[7];
    const int*   nid  = (const int*)d_in[8];

    const int N = in_sizes[0] / D_DIM;      // 100000
    const int E = in_sizes[6];              // 1000000
    const int M = 50000;                    // reference constant (not derivable from sizes)

    // workspace layout (f32 unless noted)
    char* base = (char*)d_ws;
    float* seg    = (float*)base;                                   // [M, 256]
    float* cnt    = seg + (long long)M * D_DIM;                     // [M]  (contiguous after seg)
    float* h_node = cnt + M;                                        // [N, 16]
    float* h_edge = h_node + (long long)N * H_DIM;                  // [M, 16]
    size_t tail = (((size_t)((char*)(h_edge + (long long)M * H_DIM) - base)) + 7) & ~(size_t)7;
    unsigned long long* cell = (unsigned long long*)(base + tail);  // 1 u64
    long long* sel = (long long*)(cell + 1);                        // [TOPK]

    float* out = (float*)d_out;

    // 1) zero segment buffers, reset selection
    k_init<<<2048, 256, 0, stream>>>(seg, cell, sel, (long long)M * D_DIM + M);

    // 2) segment scatter-add (E * 64 float4 chunks)
    k_scatter<<<4096, 256, 0, stream>>>(x, en, eh, seg, cnt, E);

    // 3) h_node = relu(x @ W_n + b_n)   [WMMA bf16, f32 accumulate]
    k_mlp_wmma<<<256, 256, 0, stream>>>(x, nullptr, W_n, b_n, h_node, N / 16);

    // 4) h_edge = relu((seg/cnt) @ W_e + b_e)
    k_mlp_wmma<<<128, 256, 0, stream>>>(seg, cnt, W_e, b_e, h_edge, M / 16);

    // 5) per-incidence scores -> d_out
    k_scores<<<2048, 256, 0, stream>>>(en, eh, h_node, h_edge, bias, nid, out, E);

    // 6) top-5 selection (5 argmax passes), then zero those entries
    for (int k = 0; k < TOPK; ++k) {
        k_argmax<<<1024, 256, 0, stream>>>(out, sel, cell, E);
        k_record<<<1, 1, 0, stream>>>(cell, sel, k);
    }
    k_prune<<<1, 32, 0, stream>>>(sel, out, E);
}
